// SequentialVariationalIB_36197984371159
// MI455X (gfx1250) — matrix operand order repaired and checked by reading, not compiled, not run on hardware
//
#include <hip/hip_runtime.h>
#include <hip/hip_bf16.h>
#include <stdint.h>

// ---------------------------------------------------------------------------
// Types for CDNA5 WMMA (wave32): A/B = 16 x bf16 per lane, C/D = 8 x f32.
// ---------------------------------------------------------------------------
typedef __bf16 bf16x16 __attribute__((ext_vector_type(16)));
typedef float  f32x8   __attribute__((ext_vector_type(8)));
typedef unsigned int u32x4 __attribute__((ext_vector_type(4)));

union BfPack { u32x4 q[2]; bf16x16 v; };   // 32 bytes = 16 bf16 = 8 VGPRs
static_assert(sizeof(bf16x16) == 32, "bf16x16 must be 32B");

#define VSZ 32000
#define ESZ 512
#define HSZ 512
#define BSZ 8
#define SSZ 512
#define MROWS (BSZ * SSZ)   // 4096
#define G4   (4 * HSZ)      // 2048
#define HPAD 520            // LDS row pad: 1040B stride -> 4-bank lane skew

__device__ __forceinline__ __bf16 f2bf(float f) {
  // round-to-nearest-even f32 -> bf16
  union { float f; unsigned u; } c; c.f = f;
  unsigned u = c.u;
  unsigned short s = (unsigned short)((u + 0x7FFFu + ((u >> 16) & 1u)) >> 16);
  __bf16 o;
  __builtin_memcpy(&o, &s, 2);
  return o;
}

__device__ __forceinline__ float sigm(float x) { return 1.0f / (1.0f + __expf(-x)); }

__device__ __forceinline__ f32x8 wmma_bf16(const BfPack& a, const BfPack& b, f32x8 c) {
  return __builtin_amdgcn_wmma_f32_16x16x32_bf16(false, a.v, false, b.v,
                                                 (short)0, c, false, false);
}

// ---------------------------------------------------------------------------
// Generic GEMM:  C[M,N] (f32) = A[M,K] (bf16, row-major) @ W[N,K]^T (bf16)
//                              + bias0[N] + bias1[N]   (either may be null)
// One wave per 32x32 C block: 4 accumulators, 2 A packs, 2 B packs per
// k-step -> 2x less global traffic than 16x16/wave and 4 independent WMMA
// chains for load/compute overlap. Requires M%32==0, N%32==0, K%32==0.
// ---------------------------------------------------------------------------
__global__ void __launch_bounds__(256)
gemm_bf16_wmma(const __bf16* __restrict__ A, const __bf16* __restrict__ W,
               const float* __restrict__ bias0, const float* __restrict__ bias1,
               float* __restrict__ C, int M, int N, int K) {
  const int lane = threadIdx.x & 31;
  const int wave = (blockIdx.x << 3) + (threadIdx.x >> 5);
  const int tilesN = N >> 5;
  const int tilesM = M >> 5;
  if (wave >= tilesM * tilesN) return;          // wave-uniform: EXEC stays all-1s
  const int tn = wave % tilesN;
  const int tm = wave / tilesN;
  const int m0 = tm << 5, n0 = tn << 5;
  const int ml = lane & 15, hi = lane >> 4;

  // ISA A-layout (16-bit 16x32): lane<16 -> K = {0..7, 16..23}; lane>=16 -> +8
  const __bf16* Ap0 = A + (size_t)(m0 + ml) * K + hi * 8;
  const __bf16* Ap1 = Ap0 + (size_t)16 * K;
  // ISA B-layout (16-bit 32x16): lane<16 -> K = {0..7, 16..23}; lane>=16 -> +8
  const __bf16* Wp0 = W + (size_t)(n0 + ml) * K + hi * 8;
  const __bf16* Wp1 = Wp0 + (size_t)16 * K;

  f32x8 acc00 = {}, acc01 = {}, acc10 = {}, acc11 = {};
  for (int k0 = 0; k0 < K; k0 += 32) {
    __builtin_prefetch(Wp0 + k0 + 64, 0, 1);    // global_prefetch_b8 (stream W)
    __builtin_prefetch(Wp1 + k0 + 64, 0, 1);
    BfPack a0, a1, b0, b1;
    a0.q[0] = *(const u32x4*)(Ap0 + k0); a0.q[1] = *(const u32x4*)(Ap0 + k0 + 16);
    a1.q[0] = *(const u32x4*)(Ap1 + k0); a1.q[1] = *(const u32x4*)(Ap1 + k0 + 16);
    b0.q[0] = *(const u32x4*)(Wp0 + k0); b0.q[1] = *(const u32x4*)(Wp0 + k0 + 16);
    b1.q[0] = *(const u32x4*)(Wp1 + k0); b1.q[1] = *(const u32x4*)(Wp1 + k0 + 16);
    acc00 = wmma_bf16(a0, b0, acc00);
    acc01 = wmma_bf16(a0, b1, acc01);
    acc10 = wmma_bf16(a1, b0, acc10);
    acc11 = wmma_bf16(a1, b1, acc11);
  }
  const int col0 = n0 + ml;
  const int col1 = col0 + 16;
  float bs0 = 0.0f, bs1 = 0.0f;
  if (bias0) { bs0 += bias0[col0]; bs1 += bias0[col1]; }
  if (bias1) { bs0 += bias1[col0]; bs1 += bias1[col1]; }
  // C layout: VGPR r -> row m0 + hi*8 + r, col = n0 + (lane&15)  (coalesced)
  float* C0 = C + (size_t)(m0 + hi * 8) * N + col0;   // rows m0..  , cols n0..
  float* C1 = C0 + (size_t)16 * N;                    // rows m0+16..
#pragma unroll
  for (int r = 0; r < 8; ++r) {
    C0[(size_t)r * N]      = acc00[r] + bs0;
    C0[(size_t)r * N + 16] = acc01[r] + bs1;
    C1[(size_t)r * N]      = acc10[r] + bs0;
    C1[(size_t)r * N + 16] = acc11[r] + bs1;
  }
}

// ---------------------------------------------------------------------------
// Persistent single-workgroup LSTM recurrence (one layer, all S steps).
// 32 waves; wave w owns hidden units [16w, 16w+16) across ALL four gates,
// so i/f/g/o for a unit live in that wave's accumulators (no gate shuffle).
// h (bf16, 16 x H, batch rows 8..15 = 0) lives in LDS; c lives in registers.
// gates = xproj[t] (precomputed x@W_ih.T + b_ih + b_hh) + h@W_hh.T (WMMA).
// ---------------------------------------------------------------------------
__global__ void __launch_bounds__(1024)
lstm_layer_wmma(const float* __restrict__ xproj,      // [S][B][4H] f32
                const __bf16* __restrict__ Whh,       // [4H][H] bf16
                const float* __restrict__ init_state, // (2, L, H) f32
                int layer,
                __bf16* __restrict__ hsOut) {         // [S][B][H] bf16
  __shared__ __bf16 hbuf[16][HPAD];
  const int tid  = threadIdx.x;
  const int wave = tid >> 5, lane = tid & 31;
  const int ml = lane & 15, hi = lane >> 4;
  const int u0 = wave << 4;       // 32 waves * 16 = 512 = H
  const int u  = u0 + ml;

  // h init: broadcast init_state[0][layer][:] to batch rows 0..7; rows 8..15 zero
  for (int idx = tid; idx < 16 * HSZ; idx += 1024) {
    int rowm = idx >> 9, k = idx & (HSZ - 1);
    float hv = (rowm < BSZ) ? init_state[(size_t)layer * HSZ + k] : 0.0f;
    hbuf[rowm][k] = f2bf(hv);
  }
  // c init: init_state[1][layer][u], same for every batch row
  float c_reg[8];
  {
    float c0 = init_state[(size_t)(2 + layer) * HSZ + u];
#pragma unroll
    for (int r = 0; r < 8; ++r) c_reg[r] = c0;
  }
  __syncthreads();

  // W_hh row pointers for this wave's i/f/g/o tiles (gate g rows g*H + u0 ..)
  const __bf16* W0 = Whh + (size_t)(0 * HSZ + u0 + ml) * HSZ + hi * 8;
  const __bf16* W1 = Whh + (size_t)(1 * HSZ + u0 + ml) * HSZ + hi * 8;
  const __bf16* W2 = Whh + (size_t)(2 * HSZ + u0 + ml) * HSZ + hi * 8;
  const __bf16* W3 = Whh + (size_t)(3 * HSZ + u0 + ml) * HSZ + hi * 8;

  for (int t = 0; t < SSZ; ++t) {
    f32x8 acc0 = {}, acc1 = {}, acc2 = {}, acc3 = {};
    const __bf16* Ap = &hbuf[ml][hi * 8];
    for (int k0 = 0; k0 < HSZ; k0 += 32) {
      BfPack a;
      a.q[0] = *(const u32x4*)(Ap + k0);        // ds_load_b128 from LDS
      a.q[1] = *(const u32x4*)(Ap + k0 + 16);
      BfPack b0, b1, b2, b3;
      b0.q[0] = *(const u32x4*)(W0 + k0); b0.q[1] = *(const u32x4*)(W0 + k0 + 16);
      b1.q[0] = *(const u32x4*)(W1 + k0); b1.q[1] = *(const u32x4*)(W1 + k0 + 16);
      b2.q[0] = *(const u32x4*)(W2 + k0); b2.q[1] = *(const u32x4*)(W2 + k0 + 16);
      b3.q[0] = *(const u32x4*)(W3 + k0); b3.q[1] = *(const u32x4*)(W3 + k0 + 16);
      acc0 = wmma_bf16(a, b0, acc0);
      acc1 = wmma_bf16(a, b1, acc1);
      acc2 = wmma_bf16(a, b2, acc2);
      acc3 = wmma_bf16(a, b3, acc3);
    }
    // Gate nonlinearity, fully in registers. VGPR r = batch row r (lanes<16).
    const float* xp = xproj + (size_t)t * (BSZ * G4);
    float hv[8];
#pragma unroll
    for (int r = 0; r < 8; ++r) {
      float iv = acc0[r] + xp[(size_t)r * G4 + 0 * HSZ + u];
      float fv = acc1[r] + xp[(size_t)r * G4 + 1 * HSZ + u];
      float gv = acc2[r] + xp[(size_t)r * G4 + 2 * HSZ + u];
      float ov = acc3[r] + xp[(size_t)r * G4 + 3 * HSZ + u];
      float cn = sigm(fv) * c_reg[r] + sigm(iv) * tanhf(gv);
      c_reg[r] = cn;
      hv[r] = sigm(ov) * tanhf(cn);
    }
    __syncthreads();                       // all waves done reading hbuf
    if (hi == 0) {                         // lanes 0..15 hold valid (b=r, u) data
#pragma unroll
      for (int r = 0; r < 8; ++r) {
        __bf16 hb = f2bf(hv[r]);
        hbuf[r][u] = hb;
        hsOut[((size_t)t * BSZ + r) * HSZ + u] = hb;
      }
    }
    __syncthreads();                       // h visible before next step
  }
}

// ---------------------------------------------------------------------------
// Embedding gather, directly into (S,B,E) order, bf16.
// ---------------------------------------------------------------------------
__global__ void __launch_bounds__(256)
embed_gather(const float* __restrict__ emb, const int* __restrict__ x,
             __bf16* __restrict__ out) {
  int idx = blockIdx.x * 256 + threadIdx.x;       // over S*B*E
  if (idx >= MROWS * ESZ) return;
  int e = idx & (ESZ - 1);
  int row = idx >> 9;                             // t*B + b
  int t = row >> 3, b = row & 7;
  int tok = x[b * SSZ + t];
  out[idx] = f2bf(emb[(size_t)tok * ESZ + e]);
}

__global__ void __launch_bounds__(256)
cast_f32_bf16(const float* __restrict__ in, __bf16* __restrict__ out, int n) {
  int i = blockIdx.x * 256 + threadIdx.x;
  if (i < n) out[i] = f2bf(in[i]);
}

// ---------------------------------------------------------------------------
// std = softplus(stats[:, :H]); mu = stats[:, H:]; sample = mu + sqrt(std)*eps.
// Reorders (t,b) rows -> (b,s) so the decoder GEMM writes logits directly.
// ---------------------------------------------------------------------------
__global__ void __launch_bounds__(256)
stats_sample(const float* __restrict__ stats, const float* __restrict__ eps,
             float* __restrict__ muOut, float* __restrict__ stdOut,
             __bf16* __restrict__ sampleBf) {
  int idx = blockIdx.x * 256 + threadIdx.x;       // over B*S*H
  if (idx >= MROWS * HSZ) return;
  int uu = idx & (HSZ - 1);
  int row = idx >> 9;                             // t*B + b
  int t = row >> 3, b = row & 7;
  float sraw = stats[(size_t)row * (2 * HSZ) + uu];
  float mu   = stats[(size_t)row * (2 * HSZ) + HSZ + uu];
  float sp = (sraw > 30.0f) ? sraw : log1pf(__expf(sraw));
  size_t o = ((size_t)b * SSZ + t) * HSZ + uu;    // (b, s, u)
  muOut[o]  = mu;
  stdOut[o] = sp;
  sampleBf[o] = f2bf(mu + sqrtf(sp) * eps[o]);
}

// ---------------------------------------------------------------------------
extern "C" void kernel_launch(void* const* d_in, const int* in_sizes, int n_in,
                              void* d_out, int out_size, void* d_ws, size_t ws_size,
                              hipStream_t stream) {
  (void)in_sizes; (void)n_in; (void)out_size; (void)ws_size;
  const float* emb    = (const float*)d_in[0];
  const float* istate = (const float*)d_in[1];
  const float* w_ih0  = (const float*)d_in[2];
  const float* w_hh0  = (const float*)d_in[3];
  const float* b_ih0  = (const float*)d_in[4];
  const float* b_hh0  = (const float*)d_in[5];
  const float* w_ih1  = (const float*)d_in[6];
  const float* w_hh1  = (const float*)d_in[7];
  const float* b_ih1  = (const float*)d_in[8];
  const float* b_hh1  = (const float*)d_in[9];
  const float* wg     = (const float*)d_in[10];
  const float* bg     = (const float*)d_in[11];
  const float* wd     = (const float*)d_in[12];
  const float* bd     = (const float*)d_in[13];
  const float* eps    = (const float*)d_in[14];
  const int*   x      = (const int*)d_in[15];

  float* logits = (float*)d_out;                               // [B*S, V]
  float* muOut  = logits + (size_t)MROWS * VSZ;
  float* stdOut = muOut + (size_t)MROWS * HSZ;

  // Workspace carve-out (with reuse):
  char* ws = (char*)d_ws;
  auto alloc = [&](size_t bytes) {
    char* p = ws; ws += (bytes + 255) & ~(size_t)255; return p;
  };
  __bf16* w_ih0b = (__bf16*)alloc((size_t)G4 * ESZ * 2);
  __bf16* w_hh0b = (__bf16*)alloc((size_t)G4 * HSZ * 2);
  __bf16* w_ih1b = (__bf16*)alloc((size_t)G4 * HSZ * 2);
  __bf16* w_hh1b = (__bf16*)alloc((size_t)G4 * HSZ * 2);
  __bf16* wgb    = (__bf16*)alloc((size_t)2 * HSZ * HSZ * 2);
  __bf16* wdb    = (__bf16*)alloc((size_t)VSZ * HSZ * 2);
  float*  bufA   = (float*)alloc((size_t)MROWS * G4 * 4);      // xproj0 -> xproj1 -> stats
  __bf16* embBf  = (__bf16*)alloc((size_t)MROWS * ESZ * 2);    // emb -> sampleBf
  __bf16* hs0bf  = (__bf16*)alloc((size_t)MROWS * HSZ * 2);
  __bf16* hs1bf  = (__bf16*)alloc((size_t)MROWS * HSZ * 2);
  float*  xprojL = bufA;
  float*  statsB = bufA;
  __bf16* sampleBf = embBf;

  auto cdiv = [](long long a, long long b) { return (int)((a + b - 1) / b); };

  // 1) weight casts to bf16
  cast_f32_bf16<<<cdiv(G4 * ESZ, 256), 256, 0, stream>>>(w_ih0, w_ih0b, G4 * ESZ);
  cast_f32_bf16<<<cdiv(G4 * HSZ, 256), 256, 0, stream>>>(w_hh0, w_hh0b, G4 * HSZ);
  cast_f32_bf16<<<cdiv(G4 * HSZ, 256), 256, 0, stream>>>(w_ih1, w_ih1b, G4 * HSZ);
  cast_f32_bf16<<<cdiv(G4 * HSZ, 256), 256, 0, stream>>>(w_hh1, w_hh1b, G4 * HSZ);
  cast_f32_bf16<<<cdiv(2 * HSZ * HSZ, 256), 256, 0, stream>>>(wg, wgb, 2 * HSZ * HSZ);
  cast_f32_bf16<<<cdiv((long long)VSZ * HSZ, 256), 256, 0, stream>>>(wd, wdb, VSZ * HSZ);

  // 2) embedding gather into (S,B,E) bf16
  embed_gather<<<cdiv((long long)MROWS * ESZ, 256), 256, 0, stream>>>(emb, x, embBf);

  // 3) layer 0: xproj = emb @ w_ih0.T + (b_ih0 + b_hh0); then recurrence
  {
    int tiles = (MROWS / 32) * (G4 / 32);
    gemm_bf16_wmma<<<cdiv(tiles, 8), 256, 0, stream>>>(
        embBf, w_ih0b, b_ih0, b_hh0, xprojL, MROWS, G4, ESZ);
  }
  lstm_layer_wmma<<<1, 1024, 0, stream>>>(xprojL, w_hh0b, istate, 0, hs0bf);

  // 4) layer 1
  {
    int tiles = (MROWS / 32) * (G4 / 32);
    gemm_bf16_wmma<<<cdiv(tiles, 8), 256, 0, stream>>>(
        hs0bf, w_ih1b, b_ih1, b_hh1, xprojL, MROWS, G4, HSZ);
  }
  lstm_layer_wmma<<<1, 1024, 0, stream>>>(xprojL, w_hh1b, istate, 1, hs1bf);

  // 5) stats = hs1 @ wg.T + bg
  {
    int tiles = (MROWS / 32) * ((2 * HSZ) / 32);
    gemm_bf16_wmma<<<cdiv(tiles, 8), 256, 0, stream>>>(
        hs1bf, wgb, bg, nullptr, statsB, MROWS, 2 * HSZ, HSZ);
  }

  // 6) softplus / mu / sample (reordered to (b,s) rows)
  stats_sample<<<cdiv((long long)MROWS * HSZ, 256), 256, 0, stream>>>(
      statsB, eps, muOut, stdOut, sampleBf);

  // 7) logits = sample @ wd.T + bd   (the 134 GFLOP GEMM; ~near HBM floor in bf16)
  {
    int tiles = (MROWS / 32) * (VSZ / 32);
    gemm_bf16_wmma<<<cdiv(tiles, 8), 256, 0, stream>>>(
        sampleBf, wdb, bd, nullptr, logits, MROWS, VSZ, HSZ);
  }
}
